// SeqContrastiveLoss_20504173871499
// MI455X (gfx1250) — matrix operand order repaired and checked by reading, not compiled, not run on hardware
//
#include <hip/hip_runtime.h>
#include <math.h>

// ---------------------------------------------------------------------------
// SeqContrastiveLoss for MI455X (gfx1250, wave32, WMMA)
//   N=64, S=64, D=256  ->  NS=4096 rows/view, M=8192 feature rows
//   sim = F F^T  (34.4 GFLOP) fused with fixed-shift logsumexp (shift = 1/T,
//   valid since |sim|<=1 for unit vectors) -> sim never materialized.
//   GEMM in compensated f16 (x = hi + lo): hi*hi + hi*lo + lo*hi via
//   V_WMMA_F32_16X16X32_F16, ~fp32 accuracy at f16-WMMA throughput.
// ---------------------------------------------------------------------------

typedef __attribute__((ext_vector_type(16))) _Float16 v16h;
typedef __attribute__((ext_vector_type(8)))  _Float16 v8h;
typedef __attribute__((ext_vector_type(8)))  float    v8f;

union V16 { v16h v; v8h h[2]; };

#define NSROWS   4096
#define MTOT     8192
#define DIM      256
#define INV_T    (1.0f / 0.07f)
#define CSPLIT   4                     // column-group split (partials add)
#define ROWS_WG  128                   // 8 waves * 16 rows
#define NUM_RB   (MTOT / ROWS_WG)      // 64 row blocks
#define TILES_G  (MTOT / 16 / CSPLIT)  // 128 column tiles per group

// ---------------------------------------------------------------------------
// Kernel 1: mask non-finite -> L2 normalize -> emit f32 + f16 hi/lo split.
// One wave per row (8 rows / 256-thread block).
// ---------------------------------------------------------------------------
__global__ void __launch_bounds__(256) k_normalize(const float* __restrict__ z1,
                                                   const float* __restrict__ z2,
                                                   float* __restrict__ f32,
                                                   _Float16* __restrict__ hi,
                                                   _Float16* __restrict__ lo) {
    const int wave = threadIdx.x >> 5;
    const int lane = threadIdx.x & 31;
    const int row  = blockIdx.x * 8 + wave;

    const float* src = (row < NSROWS) ? (z1 + (size_t)row * DIM)
                                      : (z2 + (size_t)(row - NSROWS) * DIM);
    float x[8];
    float ss = 0.0f;
#pragma unroll
    for (int i = 0; i < 8; ++i) {
        float v = src[lane * 8 + i];
        v = __builtin_isfinite(v) ? v : 0.0f;
        x[i] = v;
        ss += v * v;
    }
#pragma unroll
    for (int m = 16; m >= 1; m >>= 1) ss += __shfl_xor(ss, m, 32);

    const float inv = 1.0f / fmaxf(sqrtf(ss), 1e-12f);
#pragma unroll
    for (int i = 0; i < 8; ++i) {
        const float f = x[i] * inv;
        const size_t idx = (size_t)row * DIM + lane * 8 + i;
        f32[idx] = f;
        const _Float16 h = (_Float16)f;      // hi part
        hi[idx] = h;
        lo[idx] = (_Float16)(f - (float)h);  // residual
    }
}

// ---------------------------------------------------------------------------
// Kernel 2: fused F F^T + fixed-shift exp-sum.
// Grid (NUM_RB, CSPLIT); 8 waves/WG. Wave w owns rows [rb*128+w*16, +16),
// A (hi+lo, full K=256) pinned in VGPRs; streams B column tiles from L2.
// Writes per-row partial sums  s[row][g] = sum_j exp(sim(row,j)/T - 1/T).
// ---------------------------------------------------------------------------
__global__ void __launch_bounds__(256) k_sim_lse(const _Float16* __restrict__ hi,
                                                 const _Float16* __restrict__ lo,
                                                 float* __restrict__ spart) {
    const int wave    = threadIdx.x >> 5;
    const int lane    = threadIdx.x & 31;
    const int half    = lane >> 4;     // 0: lanes 0-15, 1: lanes 16-31
    const int l15     = lane & 15;
    const int rowbase = blockIdx.x * ROWS_WG + wave * 16;
    const int g       = blockIdx.y;

    // --- preload A tiles (16 rows x 256 K), hi and lo, ISA A-layout:
    // lane n (<16): row M=n, per K-step K = {32k..32k+7, 32k+16..32k+23};
    // lane n+16 adds +8 to both chunks.
    const _Float16* arow_hi = hi + (size_t)(rowbase + l15) * DIM;
    const _Float16* arow_lo = lo + (size_t)(rowbase + l15) * DIM;
    V16 ah[8], al[8];
#pragma unroll
    for (int ks = 0; ks < 8; ++ks) {
        const int k0 = ks * 32 + half * 8;
        ah[ks].h[0] = *(const v8h*)(arow_hi + k0);
        ah[ks].h[1] = *(const v8h*)(arow_hi + k0 + 16);
        al[ks].h[0] = *(const v8h*)(arow_lo + k0);
        al[ks].h[1] = *(const v8h*)(arow_lo + k0 + 16);
    }

    float s[8];
#pragma unroll
    for (int v = 0; v < 8; ++v) s[v] = 0.0f;

    const int jt0 = g * TILES_G;
    for (int jt = jt0; jt < jt0 + TILES_G; ++jt) {
        const int colbase = jt * 16;
        // B layout: lane n -> column N=n = feature row (colbase+n);
        // per K-step K = {8h..8h+7, 8h+16..8h+23} (same order as A).
        const _Float16* brow_hi = hi + (size_t)(colbase + l15) * DIM + half * 8;
        const _Float16* brow_lo = lo + (size_t)(colbase + l15) * DIM + half * 8;

        v8f c = {};
#pragma unroll
        for (int ks = 0; ks < 8; ++ks) {
            V16 bhu, blu;
            bhu.h[0] = *(const v8h*)(brow_hi + ks * 32);
            bhu.h[1] = *(const v8h*)(brow_hi + ks * 32 + 16);
            blu.h[0] = *(const v8h*)(brow_lo + ks * 32);
            blu.h[1] = *(const v8h*)(brow_lo + ks * 32 + 16);
            const v16h bh = bhu.v;
            const v16h bl = blu.v;
            c = __builtin_amdgcn_wmma_f32_16x16x32_f16(false, ah[ks].v, false, bh,
                                                       (short)0, c, false, false);
            c = __builtin_amdgcn_wmma_f32_16x16x32_f16(false, ah[ks].v, false, bl,
                                                       (short)0, c, false, false);
            c = __builtin_amdgcn_wmma_f32_16x16x32_f16(false, al[ks].v, false, bh,
                                                       (short)0, c, false, false);
        }

        // fused fixed-shift softmax-sum; C layout: c[v] = row (v + 8*half),
        // col = lane&15. Mask the diagonal element only.
        const bool diagTile = (colbase == rowbase);
#pragma unroll
        for (int v = 0; v < 8; ++v) {
            float e = __expf(c[v] * INV_T - INV_T);
            if (diagTile && (l15 == v + half * 8)) e = 0.0f;
            s[v] += e;
        }
    }

    // reduce each row's 16 per-column partials across its 16-lane group
#pragma unroll
    for (int v = 0; v < 8; ++v) {
        float t = s[v];
        t += __shfl_xor(t, 1, 32);
        t += __shfl_xor(t, 2, 32);
        t += __shfl_xor(t, 4, 32);
        t += __shfl_xor(t, 8, 32);
        if (l15 == 0) {
            const int row = rowbase + v + half * 8;
            spart[(size_t)row * CSPLIT + g] = t;
        }
    }
}

// ---------------------------------------------------------------------------
// Kernel 3: combine partials -> lse, exact f32 positive dots, mean -> d_out.
// Single 256-thread block (work is tiny).
// ---------------------------------------------------------------------------
__global__ void __launch_bounds__(256) k_finalize(const float* __restrict__ f32,
                                                  const float* __restrict__ spart,
                                                  float* __restrict__ out) {
    __shared__ float red[256];
    const int t = threadIdx.x;
    float acc = 0.0f;

    for (int j = t; j < NSROWS; j += 256) {
        const float* a = f32 + (size_t)j * DIM;
        const float* b = f32 + (size_t)(j + NSROWS) * DIM;
        float dot = 0.0f;
#pragma unroll 4
        for (int k = 0; k < DIM; k += 4) {
            const float4 av = *(const float4*)(a + k);
            const float4 bv = *(const float4*)(b + k);
            dot += av.x * bv.x + av.y * bv.y + av.z * bv.z + av.w * bv.w;
        }
        const float pos = dot * INV_T;

        float s1 = 0.0f, s2 = 0.0f;
#pragma unroll
        for (int gg = 0; gg < CSPLIT; ++gg) {
            s1 += spart[(size_t)j * CSPLIT + gg];
            s2 += spart[(size_t)(j + NSROWS) * CSPLIT + gg];
        }
        const float lse1 = INV_T + logf(s1);
        const float lse2 = INV_T + logf(s2);
        acc += (lse1 - pos) + (lse2 - pos);
    }

    red[t] = acc;
    __syncthreads();
    for (int o = 128; o > 0; o >>= 1) {
        if (t < o) red[t] += red[t + o];
        __syncthreads();
    }
    if (t == 0) out[0] = red[0] / (float)MTOT;
}

// ---------------------------------------------------------------------------
extern "C" void kernel_launch(void* const* d_in, const int* in_sizes, int n_in,
                              void* d_out, int out_size, void* d_ws, size_t ws_size,
                              hipStream_t stream) {
    (void)in_sizes; (void)n_in; (void)out_size; (void)ws_size;

    const float* z1 = (const float*)d_in[0];
    const float* z2 = (const float*)d_in[1];
    float* out = (float*)d_out;

    // workspace layout
    char* ws = (char*)d_ws;
    float*    f32   = (float*)ws;                                // 8 MiB
    _Float16* hi    = (_Float16*)(ws + (size_t)8  * 1024 * 1024); // 4 MiB
    _Float16* lo    = (_Float16*)(ws + (size_t)12 * 1024 * 1024); // 4 MiB
    float*    spart = (float*)(ws + (size_t)16 * 1024 * 1024);    // 128 KiB

    // 1) normalize + f16 split (8 rows per 256-thread block)
    k_normalize<<<MTOT / 8, 256, 0, stream>>>(z1, z2, f32, hi, lo);

    // 2) fused WMMA GEMM + fixed-shift exp sums (64 row blocks x 4 col groups)
    k_sim_lse<<<dim3(NUM_RB, CSPLIT), 256, 0, stream>>>(hi, lo, spart);

    // 3) final reduction
    k_finalize<<<1, 256, 0, stream>>>(f32, spart, out);
}
